// SecondOrder_17360257811040
// MI455X (gfx1250) — matrix operand order repaired and checked by reading, not compiled, not run on hardware
//
#include <hip/hip_runtime.h>
#include <hip/hip_bf16.h>
#include <math.h>

typedef __attribute__((ext_vector_type(16))) _Float16 v16h;
typedef __attribute__((ext_vector_type(8)))  _Float16 v8h;
typedef __attribute__((ext_vector_type(8)))  float    v8f;
typedef __attribute__((ext_vector_type(4)))  float    v4f;

#define NB 2
#define BH 128
#define BW 128
#define HW (BH*BW)

// ---- workspace layout (bytes, all 256-aligned) ----
#define OFS_EF   0ull            // [NB,H,W,224] f16 : 14,680,064 (dead after conv1)
#define OFS_H1   14680064ull     // [NB,H,W,64]  f16 :  4,194,304 (dead after conv2)
#define OFS_XN   0ull            // [NB,H,W,128] f32 : 16,777,216 (written after conv2; overlaps EF+head of H1)
#define OFS_H2   18874368ull     // [NB,H,W,64]  f16
#define OFS_H3   23068672ull     // [NB,H,W,64]  f16
#define OFS_OFF  27262976ull     // [NB,H,W,288] f32 : 37,748,736
#define OFS_MASK 65011712ull     // [NB,H,W,144] f16 :  9,437,184
#define OFS_W1P  74448896ull     // 4*63*32*16 f16
#define OFS_W2P  74706944ull     // 4*18*32*16 f16
#define OFS_W3P  74780672ull
#define OFS_W4P  74854400ull     // 28*18*32*16 f16
#define OFS_DCNP 75370496ull     // 4*36*32*16 f16

// ---------------- weight packing into WMMA A-operand layout ----------------
// A 16x32 f16: lane L holds M=L%16; element i: K = 16*hf+i
__global__ __launch_bounds__(256) void pack_conv_w(
    const float* __restrict__ w, _Float16* __restrict__ dst,
    int Cout, int Cin, int ncs, int mtiles)
{
    int idx = blockIdx.x * 256 + threadIdx.x;
    int steps = 9 * ncs;
    int total = mtiles * steps * 32;
    if (idx >= total) return;
    int lane = idx & 31;
    int s    = (idx >> 5) % steps;
    int mt   = idx / (steps * 32);
    int j  = s / ncs;                 // kernel position ky*3+kx
    int cs = s - j * ncs;             // 32-channel step
    int hf = lane >> 4;
    int m  = mt * 16 + (lane & 15);
    v16h o = {};
#pragma unroll
    for (int i = 0; i < 16; ++i) {
        int cc = 16 * hf + i;
        int c  = cs * 32 + cc;
        float v = 0.f;
        if (m < Cout && c < Cin) v = w[((size_t)m * Cin + c) * 9 + j];
        o[i] = (_Float16)v;
    }
    *(v16h*)&dst[(size_t)idx << 4] = o;
}

// dcn weights: K = (g*9+kk)*8 + c ; src w[m][g*8+c][ky][kx]
__global__ __launch_bounds__(256) void pack_dcn_w(
    const float* __restrict__ w, _Float16* __restrict__ dst)
{
    int idx = blockIdx.x * 256 + threadIdx.x;
    if (idx >= 4 * 36 * 32) return;
    int lane = idx & 31;
    int s    = (idx >> 5) % 36;
    int mt   = idx / (36 * 32);
    int hf = lane >> 4;
    int m  = mt * 16 + (lane & 15);
    v16h o = {};
#pragma unroll
    for (int i = 0; i < 16; ++i) {
        int cc = 16 * hf + i;
        int K  = s * 32 + cc;
        int c  = K & 7;
        int gk = K >> 3;
        int g  = gk / 9, kk = gk - g * 9;
        o[i] = (_Float16)w[((size_t)m * 128 + g * 8 + c) * 9 + kk];
    }
    *(v16h*)&dst[(size_t)idx << 4] = o;
}

// ---------------- input transposes ----------------
// concat(extra_feat[192], flow1[2], flow2[2]) NCHW f32 -> NHWC f16 padded to 224
__global__ __launch_bounds__(256) void ef_pack(
    const float* __restrict__ ef, const float* __restrict__ f1,
    const float* __restrict__ f2, _Float16* __restrict__ dst)
{
    size_t idx = (size_t)blockIdx.x * 256 + threadIdx.x;
    if (idx >= (size_t)NB * HW * 224) return;
    int c = (int)(idx % 224);
    size_t p = idx / 224;
    int b = (int)(p / HW);
    int pos = (int)(p % HW);
    float v = 0.f;
    if (c < 192)      v = ef[((size_t)b * 192 + c) * HW + pos];
    else if (c < 194) v = f1[((size_t)b * 2 + (c - 192)) * HW + pos];
    else if (c < 196) v = f2[((size_t)b * 2 + (c - 194)) * HW + pos];
    dst[idx] = (_Float16)v;
}

// x NCHW f32 -> NHWC f32 (read-coalesced)
__global__ __launch_bounds__(256) void x_pack(
    const float* __restrict__ xs, float* __restrict__ dst)
{
    size_t idx = (size_t)blockIdx.x * 256 + threadIdx.x;
    if (idx >= (size_t)NB * HW * 128) return;
    size_t pos = idx % HW;
    int c = (int)((idx / HW) & 127);
    int b = (int)(idx / ((size_t)HW * 128));
    dst[((size_t)b * HW + pos) * 128 + c] = xs[idx];
}

// ---------------- generic 3x3 conv (Cout=64) + leakyReLU, WMMA ----------------
__global__ __launch_bounds__(128) void conv3x3_wmma_leaky(
    const _Float16* __restrict__ in_nhwc,  // [NB,H,W,Cinp]
    const _Float16* __restrict__ apack,    // [4][9*ncs][32][16]
    const float* __restrict__ bias,        // [64]
    _Float16* __restrict__ out_nhwc,       // [NB,H,W,64]
    int Cinp, int ncs)
{
    __shared__ __attribute__((aligned(32))) _Float16 patch[3 * 18 * 224];
    const int tid  = threadIdx.x;
    const int tile = blockIdx.x;
    const int x0 = (tile & 7) << 4;
    const int y  = (tile >> 3) & (BH - 1);
    const int b  = tile >> 10;

    const int chunks = Cinp >> 3;
    const int total  = 54 * chunks;
    for (int idx = tid; idx < total; idx += 128) {
        int site = idx / chunks;
        int ck   = idx - site * chunks;
        int r = site / 18, colp = site - r * 18;
        int yy = y + r - 1, xx = x0 + colp - 1;
        v8h v = {};
        if (yy >= 0 && yy < BH && xx >= 0 && xx < BW)
            v = *(const v8h*)&in_nhwc[((size_t)((b * BH + yy) * BW + xx)) * Cinp + (ck << 3)];
        *(v8h*)&patch[site * Cinp + (ck << 3)] = v;
    }
    __syncthreads();

    const int wave = tid >> 5, lane = tid & 31;
    const int col = lane & 15, hf = lane >> 4;
    const int steps = 9 * ncs;
    v8f acc = {};
    for (int j = 0; j < 9; ++j) {
        int ky = j / 3, kx = j - ky * 3;
        int baseB = (ky * 18 + kx + col) * Cinp + 16 * hf;
        for (int cs = 0; cs < ncs; ++cs) {
            v16h bv = *(const v16h*)&patch[baseB + (cs << 5)];
            int s = j * ncs + cs;
            v16h av = *(const v16h*)&apack[((size_t)(wave * steps + s) * 32 + lane) << 4];
            acc = __builtin_amdgcn_wmma_f32_16x16x32_f16(
                false, av, false, bv, (short)0, acc, false, false);
        }
    }
    const int base = wave * 16 + 8 * hf;
    const int px = x0 + col;
    v8h o;
#pragma unroll
    for (int r = 0; r < 8; ++r) {
        float v = acc[r] + bias[base + r];
        o[r] = (_Float16)(v >= 0.f ? v : 0.1f * v);
    }
    *(v8h*)&out_nhwc[((size_t)((b * BH + y) * BW + px)) * 64 + base] = o;
}

// ------------- conv4: 64 -> 432, fused offset/mask epilogue -------------
__global__ __launch_bounds__(128) void conv4_wmma(
    const _Float16* __restrict__ in_nhwc,  // [NB,H,W,64]
    const _Float16* __restrict__ apack,    // [28][18][32][16]
    const float* __restrict__ bias,        // [432]
    const float* __restrict__ f1, const float* __restrict__ f2,
    float* __restrict__ offb,              // [NB,H,W,288] f32
    _Float16* __restrict__ maskb)          // [NB,H,W,144] f16
{
    __shared__ __attribute__((aligned(32))) _Float16 patch[3 * 18 * 64];
    const int tid  = threadIdx.x;
    const int tile = blockIdx.x;
    const int x0 = (tile & 7) << 4;
    const int y  = (tile >> 3) & (BH - 1);
    const int b  = tile >> 10;

    for (int idx = tid; idx < 54 * 8; idx += 128) {
        int site = idx >> 3;
        int ck   = idx & 7;
        int r = site / 18, colp = site - r * 18;
        int yy = y + r - 1, xx = x0 + colp - 1;
        v8h v = {};
        if (yy >= 0 && yy < BH && xx >= 0 && xx < BW)
            v = *(const v8h*)&in_nhwc[((size_t)((b * BH + yy) * BW + xx)) * 64 + (ck << 3)];
        *(v8h*)&patch[site * 64 + (ck << 3)] = v;
    }
    __syncthreads();

    const int wave = tid >> 5, lane = tid & 31;
    const int col = lane & 15, hf = lane >> 4;
    const int mtile = blockIdx.y * 4 + wave;
    v8f acc = {};
    for (int j = 0; j < 9; ++j) {
        int ky = j / 3, kx = j - ky * 3;
        int baseB = (ky * 18 + kx + col) * 64 + 16 * hf;
        for (int cs = 0; cs < 2; ++cs) {
            v16h bv = *(const v16h*)&patch[baseB + (cs << 5)];
            int s = j * 2 + cs;
            v16h av = *(const v16h*)&apack[((size_t)(mtile * 18 + s) * 32 + lane) << 4];
            acc = __builtin_amdgcn_wmma_f32_16x16x32_f16(
                false, av, false, bv, (short)0, acc, false, false);
        }
    }
    if (mtile >= 27) return;
    const int base = mtile * 16 + 8 * hf;       // 0..424, multiple of 8
    const int px = x0 + col;
    const size_t p = (size_t)((b * BH + y) * BW + px);
    if (base < 288) {
        // offset = 10*tanh(.) + tiled flipped flow
        float f1y = f1[((size_t)b * 2 + 1) * HW + y * BW + px];
        float f1x = f1[((size_t)b * 2 + 0) * HW + y * BW + px];
        float f2y = f2[((size_t)b * 2 + 1) * HW + y * BW + px];
        float f2x = f2[((size_t)b * 2 + 0) * HW + y * BW + px];
        v4f o0, o1;
#pragma unroll
        for (int r = 0; r < 8; ++r) {
            int ch = base + r;
            float v = 10.f * tanhf(acc[r] + bias[ch]);
            float fl;
            if (ch < 144) fl = ((ch & 1) == 0) ? f1y : f1x;
            else          fl = ((ch & 1) == 0) ? f2y : f2x;  // 144 is even
            if (r < 4) o0[r] = v + fl; else o1[r - 4] = v + fl;
        }
        *(v4f*)&offb[p * 288 + base]     = o0;
        *(v4f*)&offb[p * 288 + base + 4] = o1;
    } else {
        v8h mo;
#pragma unroll
        for (int r = 0; r < 8; ++r) {
            float v = acc[r] + bias[base + r];
            mo[r] = (_Float16)(1.f / (1.f + expf(-v)));
        }
        *(v8h*)&maskb[p * 144 + (base - 288)] = mo;
    }
}

// ------------- modulated deformable conv, WMMA GEMM over K=1152 -------------
__global__ __launch_bounds__(128) void dcn_wmma(
    const float* __restrict__ xn,        // [NB,H,W,128] f32
    const float* __restrict__ offb,      // [NB,H,W,288] f32
    const _Float16* __restrict__ maskb,  // [NB,H,W,144] f16
    const _Float16* __restrict__ apack,  // [4][36][32][16]
    const float* __restrict__ dbias,     // [64]
    float* __restrict__ out)             // [NB,64,H,W] f32
{
    __shared__ __attribute__((aligned(32))) _Float16 val[16 * 1152];
    const int tid  = threadIdx.x;
    const int tile = blockIdx.x;
    const int x0 = (tile & 7) << 4;
    const int y  = (tile >> 3) & (BH - 1);
    const int b  = tile >> 10;

    // sampling: 16 pixels x 144 (g,k) tasks, 8 channels each
    for (int t = tid; t < 2304; t += 128) {
        int pix = t / 144;
        int gk  = t - pix * 144;
        int g = gk / 9, kk = gk - g * 9;
        int px = x0 + pix;
        size_t p = (size_t)((b * BH + y) * BW + px);
        float oy = offb[p * 288 + gk * 2 + 0];
        float ox = offb[p * 288 + gk * 2 + 1];
        float m  = (float)maskb[p * 144 + gk];
        float py  = (float)(y + (kk / 3) - 1) + oy;
        float pxf = (float)(px + (kk - (kk / 3) * 3) - 1) + ox;
        float y0f = floorf(py), x0f = floorf(pxf);
        float wy = py - y0f, wx = pxf - x0f;
        int iy = (int)y0f, ix = (int)x0f;
        float a0 = 0.f, a1 = 0.f, a2 = 0.f, a3 = 0.f, a4 = 0.f, a5 = 0.f, a6 = 0.f, a7 = 0.f;
#pragma unroll
        for (int cn = 0; cn < 4; ++cn) {
            int yy = iy + (cn >> 1), xx = ix + (cn & 1);
            float wgt = ((cn >> 1) ? wy : 1.f - wy) * ((cn & 1) ? wx : 1.f - wx);
            if (yy >= 0 && yy < BH && xx >= 0 && xx < BW) {
                const float* s = &xn[((size_t)((b * BH + yy) * BW + xx)) * 128 + g * 8];
                v4f lo = *(const v4f*)s;
                v4f hi = *(const v4f*)(s + 4);
                a0 += wgt * lo[0]; a1 += wgt * lo[1]; a2 += wgt * lo[2]; a3 += wgt * lo[3];
                a4 += wgt * hi[0]; a5 += wgt * hi[1]; a6 += wgt * hi[2]; a7 += wgt * hi[3];
            }
        }
        v8h o;
        o[0] = (_Float16)(m * a0); o[1] = (_Float16)(m * a1);
        o[2] = (_Float16)(m * a2); o[3] = (_Float16)(m * a3);
        o[4] = (_Float16)(m * a4); o[5] = (_Float16)(m * a5);
        o[6] = (_Float16)(m * a6); o[7] = (_Float16)(m * a7);
        *(v8h*)&val[pix * 1152 + gk * 8] = o;
    }
    __syncthreads();

    const int wave = tid >> 5, lane = tid & 31;
    const int col = lane & 15, hf = lane >> 4;
    v8f acc = {};
    for (int s = 0; s < 36; ++s) {
        v16h bv = *(const v16h*)&val[col * 1152 + (s << 5) + 16 * hf];
        v16h av = *(const v16h*)&apack[((size_t)(wave * 36 + s) * 32 + lane) << 4];
        acc = __builtin_amdgcn_wmma_f32_16x16x32_f16(
            false, av, false, bv, (short)0, acc, false, false);
    }
    const int base = wave * 16 + 8 * hf;
    const int px = x0 + col;
#pragma unroll
    for (int r = 0; r < 8; ++r) {
        int ch = base + r;
        out[((size_t)(b * 64 + ch)) * HW + y * BW + px] = acc[r] + dbias[ch];
    }
}

// ---------------------------------------------------------------------------
extern "C" void kernel_launch(void* const* d_in, const int* in_sizes, int n_in,
                              void* d_out, int out_size, void* d_ws, size_t ws_size,
                              hipStream_t stream) {
    const float* x    = (const float*)d_in[0];
    const float* ef   = (const float*)d_in[1];
    const float* f1   = (const float*)d_in[2];
    const float* f2   = (const float*)d_in[3];
    const float* w1   = (const float*)d_in[4];
    const float* b1   = (const float*)d_in[5];
    const float* w2   = (const float*)d_in[6];
    const float* b2   = (const float*)d_in[7];
    const float* w3   = (const float*)d_in[8];
    const float* b3   = (const float*)d_in[9];
    const float* w4   = (const float*)d_in[10];
    const float* b4   = (const float*)d_in[11];
    const float* dcnw = (const float*)d_in[12];
    const float* dcnb = (const float*)d_in[13];

    char* ws = (char*)d_ws;
    _Float16* efn   = (_Float16*)(ws + OFS_EF);
    _Float16* h1    = (_Float16*)(ws + OFS_H1);
    float*    xn    = (float*)   (ws + OFS_XN);
    _Float16* h2    = (_Float16*)(ws + OFS_H2);
    _Float16* h3    = (_Float16*)(ws + OFS_H3);
    float*    offb  = (float*)   (ws + OFS_OFF);
    _Float16* maskb = (_Float16*)(ws + OFS_MASK);
    _Float16* w1p   = (_Float16*)(ws + OFS_W1P);
    _Float16* w2p   = (_Float16*)(ws + OFS_W2P);
    _Float16* w3p   = (_Float16*)(ws + OFS_W3P);
    _Float16* w4p   = (_Float16*)(ws + OFS_W4P);
    _Float16* dcnp  = (_Float16*)(ws + OFS_DCNP);

    // weight packing
    pack_conv_w<<<(4 * 63 * 32 + 255) / 256, 256, 0, stream>>>(w1, w1p, 64, 196, 7, 4);
    pack_conv_w<<<(4 * 18 * 32 + 255) / 256, 256, 0, stream>>>(w2, w2p, 64, 64, 2, 4);
    pack_conv_w<<<(4 * 18 * 32 + 255) / 256, 256, 0, stream>>>(w3, w3p, 64, 64, 2, 4);
    pack_conv_w<<<(28 * 18 * 32 + 255) / 256, 256, 0, stream>>>(w4, w4p, 432, 64, 2, 28);
    pack_dcn_w<<<(4 * 36 * 32 + 255) / 256, 256, 0, stream>>>(dcnw, dcnp);

    // input concat/transpose to NHWC f16 (padded to 224)
    ef_pack<<<(NB * HW * 224) / 256, 256, 0, stream>>>(ef, f1, f2, efn);

    const int ntiles = NB * BH * (BW / 16);  // 2048 pixel tiles
    conv3x3_wmma_leaky<<<ntiles, 128, 0, stream>>>(efn, w1p, b1, h1, 224, 7);
    conv3x3_wmma_leaky<<<ntiles, 128, 0, stream>>>(h1, w2p, b2, h2, 64, 2);

    // x -> NHWC f32 (placed over dead ef/h1 region: must follow conv2)
    x_pack<<<(NB * HW * 128) / 256, 256, 0, stream>>>(x, xn);

    conv3x3_wmma_leaky<<<ntiles, 128, 0, stream>>>(h2, w3p, b3, h3, 64, 2);
    conv4_wmma<<<dim3(ntiles, 7), 128, 0, stream>>>(h3, w4p, b4, f1, f2, offb, maskb);
    dcn_wmma<<<ntiles, 128, 0, stream>>>(xn, offb, maskb, dcnp, dcnb, (float*)d_out);
}
